// QuaternionLinear_65352222376532
// MI455X (gfx1250) — compile-verified, matrix operand order checked by reading, not run on hardware
//
#include <hip/hip_runtime.h>

// ---------------------------------------------------------------------------
// Quaternion linear = one GEMM  out[M,N] = X[M,K] * W^T + bias
//   M = 8192, N = 4096, K = 4096  ->  275 GFLOP over ~272 MB  (~1000 FLOP/B)
// Compute-bound on the WMMA pipes. fp32 accuracy via bf16x3 split:
//   Ahi*Bhi + Alo*Bhi + Ahi*Blo   (f32 accumulate, 3 WMMAs per K=32)
// Hamilton block structure: sel = obq ^ fbq, sign from 16-bit mask; W never
// materialized (reads the raw 16 MB of Wr/Wi/Wj/Wk).
// ---------------------------------------------------------------------------

typedef __attribute__((ext_vector_type(16))) __bf16       v16bf;
typedef __attribute__((ext_vector_type(8)))  float        v8f;
typedef __attribute__((ext_vector_type(4)))  __bf16       bf16x4;
typedef __attribute__((ext_vector_type(4)))  float        f32x4;
typedef __attribute__((ext_vector_type(4)))  unsigned int u32x4;

#define IN_F   4096
#define OUT_F  4096
#define M_TOT  8192
#define BM     128
#define BN     64
#define BK     32
#define LDK    40           // padded LDS row stride (bf16): 80 B = 20 banks
#define NKT    (IN_F / BK)  // 128 K-tiles
#define HAM_NEG_MASK 0x428E // neg blocks at idx {1,2,3,7,9,14}

__global__ __launch_bounds__(256) void qlin_wmma_bf16x3(
    const float* __restrict__ X,
    const float* __restrict__ Wr, const float* __restrict__ Wi,
    const float* __restrict__ Wj, const float* __restrict__ Wk,
    const float* __restrict__ bias,
    float* __restrict__ Out)
{
    __shared__ __bf16 sAhi[2][BM * LDK];
    __shared__ __bf16 sAlo[2][BM * LDK];
    __shared__ __bf16 sBhi[2][BN * LDK];
    __shared__ __bf16 sBlo[2][BN * LDK];

    const int tid   = threadIdx.x;
    const int lane  = tid & 31;    // wave32
    const int wave  = tid >> 5;    // 0..7
    const int waveM = wave >> 1;   // 0..3 : 32 M-rows each
    const int waveN = wave & 1;    // 0..1 : 32 N-cols each

    const int m0  = blockIdx.x * BM;
    const int n0  = blockIdx.y * BN;
    const int obq = n0 >> 10;      // Hamilton row-block (constant per CTA)
    const int oi0 = n0 & 1023;

    // global-load mapping: 8 threads x float4 cover one 32-float K-row
    const int lrow  = tid >> 3;        // 0..31
    const int lcol4 = (tid & 7) << 2;  // 0..28

    const float* __restrict__ xbase = X + (size_t)(m0 + lrow) * IN_F + lcol4;

    // LDS fragment addressing (ISA 16-bit A 16x32 lane layout), loop-invariant
    const int fr = lane & 15;
    const int fk = (lane >> 4) << 3;   // 0 or 8
    const int aoff = (waveM * 32 + fr) * LDK + fk;
    const int boff = (waveN * 32 + fr) * LDK + fk;

    f32x4 ra[4];   // staged A rows: lrow, +32, +64, +96  (raw f32)
    f32x4 rb[2];   // staged W rows: lrow, +32            (raw f32, unsigned)

    v8f acc[2][2] = {};

    // ---- prologue: issue K-tile 0 loads ----
    {
#pragma unroll
        for (int p = 0; p < 4; ++p)
            ra[p] = *(const f32x4*)(xbase + (size_t)p * 32 * IN_F);
        const float* __restrict__ Wp = (obq == 0) ? Wr : (obq == 1) ? Wi
                                     : (obq == 2) ? Wj : Wk;     // fbq = 0
#pragma unroll
        for (int p = 0; p < 2; ++p)
            rb[p] = *(const f32x4*)(Wp + (size_t)(oi0 + lrow + p * 32) * 1024 + lcol4);
    }

    for (int kt = 0; kt < NKT; ++kt) {
        const int buf = kt & 1;

        // ---- convert staged tile kt -> bf16 hi/lo, store into LDS[buf] ----
        // (sign of the Hamilton block folded in here, NOT at load time, so the
        //  global loads issued last iteration are only waited on now)
        {
            const int   fbq = (kt * BK) >> 10;
            const float sgn = ((HAM_NEG_MASK >> (obq * 4 + fbq)) & 1) ? -1.0f : 1.0f;
#pragma unroll
            for (int p = 0; p < 4; ++p) {
                bf16x4 h, l;
#pragma unroll
                for (int e = 0; e < 4; ++e) {
                    const float  v  = ra[p][e];
                    const __bf16 hb = (__bf16)v;
                    h[e] = hb;
                    l[e] = (__bf16)(v - (float)hb);
                }
                const int r = lrow + p * 32;
                *(bf16x4*)&sAhi[buf][r * LDK + lcol4] = h;
                *(bf16x4*)&sAlo[buf][r * LDK + lcol4] = l;
            }
#pragma unroll
            for (int p = 0; p < 2; ++p) {
                bf16x4 h, l;
#pragma unroll
                for (int e = 0; e < 4; ++e) {
                    const float  v  = rb[p][e] * sgn;
                    const __bf16 hb = (__bf16)v;
                    h[e] = hb;
                    l[e] = (__bf16)(v - (float)hb);
                }
                const int r = lrow + p * 32;
                *(bf16x4*)&sBhi[buf][r * LDK + lcol4] = h;
                *(bf16x4*)&sBlo[buf][r * LDK + lcol4] = l;
            }
        }
        __syncthreads();   // one barrier per K-iteration is sufficient

        // ---- issue tile kt+1 global loads; latency hidden by WMMAs below ----
        if (kt + 1 < NKT) {
            const int k = (kt + 1) * BK;
#pragma unroll
            for (int p = 0; p < 4; ++p)
                ra[p] = *(const f32x4*)(xbase + (size_t)p * 32 * IN_F + k);
            const int fbq = k >> 10;
            const int fi0 = k & 1023;
            const int sel = obq ^ fbq;   // Hamilton XOR table
            const float* __restrict__ Wp = (sel == 0) ? Wr : (sel == 1) ? Wi
                                         : (sel == 2) ? Wj : Wk;
#pragma unroll
            for (int p = 0; p < 2; ++p)
                rb[p] = *(const f32x4*)(Wp + (size_t)(oi0 + lrow + p * 32) * 1024
                                        + fi0 + lcol4);
            if (kt + 2 < NKT)
                __builtin_prefetch(xbase + (size_t)(kt + 2) * BK, 0, 1);
        }

        // ---- WMMA on LDS[buf]: A-frags held, B-frags streamed per j ----
        {
            const __bf16* pAhi = &sAhi[buf][aoff];
            const __bf16* pAlo = &sAlo[buf][aoff];
            const __bf16* pBhi = &sBhi[buf][boff];
            const __bf16* pBlo = &sBlo[buf][boff];

            v16bf ah[2], al[2];
#pragma unroll
            for (int i = 0; i < 2; ++i) {
                ((u32x4*)&ah[i])[0] = *(const u32x4*)(pAhi + i * 16 * LDK);
                ((u32x4*)&ah[i])[1] = *(const u32x4*)(pAhi + i * 16 * LDK + 16);
                ((u32x4*)&al[i])[0] = *(const u32x4*)(pAlo + i * 16 * LDK);
                ((u32x4*)&al[i])[1] = *(const u32x4*)(pAlo + i * 16 * LDK + 16);
            }
#pragma unroll
            for (int j = 0; j < 2; ++j) {
                v16bf bh, bl;
                ((u32x4*)&bh)[0] = *(const u32x4*)(pBhi + j * 16 * LDK);
                ((u32x4*)&bh)[1] = *(const u32x4*)(pBhi + j * 16 * LDK + 16);
                ((u32x4*)&bl)[0] = *(const u32x4*)(pBlo + j * 16 * LDK);
                ((u32x4*)&bl)[1] = *(const u32x4*)(pBlo + j * 16 * LDK + 16);
#pragma unroll
                for (int i = 0; i < 2; ++i) {
                    v8f c = acc[i][j];
                    c = __builtin_amdgcn_wmma_f32_16x16x32_bf16(
                            false, ah[i], false, bh, (short)0, c, false, false);
                    c = __builtin_amdgcn_wmma_f32_16x16x32_bf16(
                            false, al[i], false, bh, (short)0, c, false, false);
                    c = __builtin_amdgcn_wmma_f32_16x16x32_bf16(
                            false, ah[i], false, bl, (short)0, c, false, false);
                    acc[i][j] = c;
                }
            }
        }
    }

    // ---- epilogue: f32 16x16 C/D layout; non-temporal (write-once stream) ----
    //   lanes 0-15: M=r, N=lane;  lanes 16-31: M=r+8, N=lane-16
    const int mofs = (lane >> 4) << 3;
    const int coll = lane & 15;
#pragma unroll
    for (int i = 0; i < 2; ++i) {
#pragma unroll
        for (int j = 0; j < 2; ++j) {
            const int row0 = m0 + waveM * 32 + i * 16 + mofs;
            const int col  = n0 + waveN * 32 + j * 16 + coll;
            const float bv = bias[col];
#pragma unroll
            for (int r = 0; r < 8; ++r) {
                __builtin_nontemporal_store(acc[i][j][r] + bv,
                                            Out + (size_t)(row0 + r) * OUT_F + col);
            }
        }
    }
}

extern "C" void kernel_launch(void* const* d_in, const int* in_sizes, int n_in,
                              void* d_out, int out_size, void* d_ws, size_t ws_size,
                              hipStream_t stream) {
    (void)in_sizes; (void)n_in; (void)out_size; (void)d_ws; (void)ws_size;
    const float* X    = (const float*)d_in[0];
    const float* Wr   = (const float*)d_in[1];
    const float* Wi   = (const float*)d_in[2];
    const float* Wj   = (const float*)d_in[3];
    const float* Wk   = (const float*)d_in[4];
    const float* bias = (const float*)d_in[5];
    float* Out = (float*)d_out;

    dim3 grid(M_TOT / BM, OUT_F / BN);  // 64 x 64 CTAs
    dim3 block(256);                    // 8 wave32 waves
    qlin_wmma_bf16x3<<<grid, block, 0, stream>>>(X, Wr, Wi, Wj, Wk, bias, Out);
}
